// Sinkhorn_7335804141998
// MI455X (gfx1250) — matrix operand order repaired and checked by reading, not compiled, not run on hardware
//
#include <hip/hip_runtime.h>
#include <stdint.h>

// Scaling-form Sinkhorn:  u = alpha / (K v),  v = beta / (K^T u),  K = exp(-C/eps)
//   f = eps*log(u), g = eps*log(v)
// K is precomputed once in bf16 (128 MB -> resident in MI455X 192 MB L2);
// all 32 GEMV sweeps then run at L2 bandwidth. Row GEMV uses
// v_wmma_f32_16x16x32_bf16 (16 rows x 32 K-slice per instruction).

#define NA 8192
#define NB 8192
#define LAYERS 16

typedef __attribute__((ext_vector_type(16))) __bf16 v16bf;
typedef __attribute__((ext_vector_type(8)))  float  v8f;

__device__ __forceinline__ unsigned short f32_to_bf16(float x) {
    union { float f; uint32_t u; } c; c.f = x;
    uint32_t r = c.u + 0x7FFFu + ((c.u >> 16) & 1u);   // round-nearest-even
    return (unsigned short)(r >> 16);
}
__device__ __forceinline__ float bflo(uint32_t x) {
    union { uint32_t u; float f; } c; c.u = x << 16; return c.f;
}
__device__ __forceinline__ float bfhi(uint32_t x) {
    union { uint32_t u; float f; } c; c.u = x & 0xFFFF0000u; return c.f;
}

// ---------------------------------------------------------------- K = exp(-C/eps)
__global__ void build_k(const float* __restrict__ C, const float* __restrict__ epsp,
                        unsigned short* __restrict__ Kb) {
    const float ninv_eps = -1.0f / epsp[0];
    size_t idx    = ((size_t)blockIdx.x * blockDim.x + threadIdx.x) * 8;
    const size_t total  = (size_t)NA * NB;
    const size_t stride = (size_t)gridDim.x * blockDim.x * 8;
    for (; idx < total; idx += stride) {
        float4 a = *(const float4*)(C + idx);
        float4 b = *(const float4*)(C + idx + 4);
        uint32_t p0 = (uint32_t)f32_to_bf16(__expf(a.x * ninv_eps)) |
                      ((uint32_t)f32_to_bf16(__expf(a.y * ninv_eps)) << 16);
        uint32_t p1 = (uint32_t)f32_to_bf16(__expf(a.z * ninv_eps)) |
                      ((uint32_t)f32_to_bf16(__expf(a.w * ninv_eps)) << 16);
        uint32_t p2 = (uint32_t)f32_to_bf16(__expf(b.x * ninv_eps)) |
                      ((uint32_t)f32_to_bf16(__expf(b.y * ninv_eps)) << 16);
        uint32_t p3 = (uint32_t)f32_to_bf16(__expf(b.z * ninv_eps)) |
                      ((uint32_t)f32_to_bf16(__expf(b.w * ninv_eps)) << 16);
        uint4 o; o.x = p0; o.y = p1; o.z = p2; o.w = p3;
        *(uint4*)(Kb + idx) = o;
    }
}

// ---------------------------------------------------------------- init: v = 1, sums = 0
__global__ void init_uv(float* __restrict__ v, unsigned short* __restrict__ vb,
                        float* __restrict__ rowsum, float* __restrict__ colsum) {
    int i = blockIdx.x * blockDim.x + threadIdx.x;
    if (i < NB) {
        v[i] = 1.0f;
        vb[i] = 0x3F80;         // bf16(1.0)
        rowsum[i] = 0.0f;
        colsum[i] = 0.0f;
    }
}

// ---------------------------------------------------------------- rowsum[i] = sum_j K[i,j]*v[j]  (WMMA)
// One wave handles a 16-row tile over a 1024-wide j-chunk. B is the v-chunk
// replicated into every column, so each D column holds the 16 row partial sums.
__global__ __launch_bounds__(256) void row_pass(const unsigned short* __restrict__ Kb,
                                                const unsigned short* __restrict__ vb,
                                                float* __restrict__ rowsum) {
    const int lane = threadIdx.x & 31;
    const int wave = threadIdx.x >> 5;
    const int tile = blockIdx.x * 8 + wave;      // 512 row tiles total
    const int row0 = tile * 16;
    const int jbase = blockIdx.y * (NB / 8);     // 8 chunks of 1024
    const int row  = row0 + (lane & 15);
    const int aoff = (lane < 16) ? 0 : 8;        // A: lanes 0-15 hold K {0..7,16..23}
    const int boff = (lane < 16) ? 0 : 8;        // B: lanes 0-15 hold K {0..7,16..23}
    const unsigned short* arow = Kb + (size_t)row * NB;

    v8f acc = {};
    for (int j = jbase; j < jbase + NB / 8; j += 32) {
        union { v16bf v; uint4 q[2]; } A, B;
        A.q[0] = *(const uint4*)(arow + j + aoff);
        A.q[1] = *(const uint4*)(arow + j + aoff + 16);
        B.q[0] = *(const uint4*)(vb + j + boff);
        B.q[1] = *(const uint4*)(vb + j + boff + 16);
        acc = __builtin_amdgcn_wmma_f32_16x16x32_bf16(false, A.v, false, B.v,
                                                      (short)0, acc, false, false);
    }
    // D layout: VGPR r, lanes 0-15 -> row r; lanes 16-31 -> row r+8 (all cols equal)
    if (lane == 0) {
#pragma unroll
        for (int r = 0; r < 8; ++r) atomicAdd(&rowsum[row0 + r], acc[r]);
    } else if (lane == 16) {
#pragma unroll
        for (int r = 0; r < 8; ++r) atomicAdd(&rowsum[row0 + 8 + r], acc[r]);
    }
}

// ---------------------------------------------------------------- u = alpha/rowsum ; zero colsum
__global__ void finish_u(const float* __restrict__ alpha, const float* __restrict__ rowsum,
                         float* __restrict__ u, float* __restrict__ colsum) {
    int i = blockIdx.x * blockDim.x + threadIdx.x;
    if (i < NA) {
        u[i] = alpha[i] / rowsum[i];
        colsum[i] = 0.0f;
    }
}

// ---------------------------------------------------------------- colsum[j] = sum_i K[i,j]*u[i]
// Row-major coalesced streaming: each thread owns 8 columns; u[i] is a uniform
// scalar load. K is L2-resident bf16, so this sweep also runs at L2 bandwidth.
__global__ __launch_bounds__(256) void col_pass(const unsigned short* __restrict__ Kb,
                                                const float* __restrict__ u,
                                                float* __restrict__ colsum) {
    const int col = blockIdx.x * 2048 + threadIdx.x * 8;   // 4 col-blocks
    const int i0  = blockIdx.y * (NA / 32);                // 32 chunks of 256 rows
    float a0 = 0.f, a1 = 0.f, a2 = 0.f, a3 = 0.f, a4 = 0.f, a5 = 0.f, a6 = 0.f, a7 = 0.f;
    for (int i = i0; i < i0 + NA / 32; ++i) {
        const unsigned short* p = Kb + (size_t)i * NB + col;
        __builtin_prefetch(p + 4 * NB, 0, 1);              // global_prefetch_b8
        const uint4 kq = *(const uint4*)p;
        const float ui = u[i];
        a0 = fmaf(bflo(kq.x), ui, a0); a1 = fmaf(bfhi(kq.x), ui, a1);
        a2 = fmaf(bflo(kq.y), ui, a2); a3 = fmaf(bfhi(kq.y), ui, a3);
        a4 = fmaf(bflo(kq.z), ui, a4); a5 = fmaf(bfhi(kq.z), ui, a5);
        a6 = fmaf(bflo(kq.w), ui, a6); a7 = fmaf(bfhi(kq.w), ui, a7);
    }
    atomicAdd(&colsum[col + 0], a0); atomicAdd(&colsum[col + 1], a1);
    atomicAdd(&colsum[col + 2], a2); atomicAdd(&colsum[col + 3], a3);
    atomicAdd(&colsum[col + 4], a4); atomicAdd(&colsum[col + 5], a5);
    atomicAdd(&colsum[col + 6], a6); atomicAdd(&colsum[col + 7], a7);
}

// ---------------------------------------------------------------- v = beta/colsum ; zero rowsum
__global__ void finish_v(const float* __restrict__ beta, const float* __restrict__ colsum,
                         float* __restrict__ v, unsigned short* __restrict__ vb,
                         float* __restrict__ rowsum) {
    int j = blockIdx.x * blockDim.x + threadIdx.x;
    if (j < NB) {
        float vj = beta[j] / colsum[j];
        v[j] = vj;
        vb[j] = f32_to_bf16(vj);
        rowsum[j] = 0.0f;
    }
}

// ---------------------------------------------------------------- f = eps*log(u), g = eps*log(v)
__global__ void finalize(const float* __restrict__ u, const float* __restrict__ v,
                         const float* __restrict__ epsp, float* __restrict__ out) {
    int i = blockIdx.x * blockDim.x + threadIdx.x;
    if (i < NA) {
        const float e = epsp[0];
        out[i]       = e * __logf(u[i]);
        out[NA + i]  = e * __logf(v[i]);
    }
}

extern "C" void kernel_launch(void* const* d_in, const int* in_sizes, int n_in,
                              void* d_out, int out_size, void* d_ws, size_t ws_size,
                              hipStream_t stream) {
    const float* alpha = (const float*)d_in[0];
    const float* beta  = (const float*)d_in[1];
    const float* C     = (const float*)d_in[2];
    const float* eps   = (const float*)d_in[3];
    float* out = (float*)d_out;

    uint8_t* ws = (uint8_t*)d_ws;
    size_t off = 0;
    unsigned short* Kb = (unsigned short*)(ws + off); off += (size_t)NA * NB * 2; // 128 MB
    float* u      = (float*)(ws + off); off += NA * sizeof(float);
    float* v      = (float*)(ws + off); off += NB * sizeof(float);
    float* rowsum = (float*)(ws + off); off += NA * sizeof(float);
    float* colsum = (float*)(ws + off); off += NB * sizeof(float);
    unsigned short* vb = (unsigned short*)(ws + off); off += NB * 2;

    build_k<<<dim3(16384), dim3(256), 0, stream>>>(C, eps, Kb);
    init_uv<<<dim3(32), dim3(256), 0, stream>>>(v, vb, rowsum, colsum);

    for (int t = 0; t < LAYERS; ++t) {
        row_pass<<<dim3(64, 8), dim3(256), 0, stream>>>(Kb, vb, rowsum);
        finish_u<<<dim3(32), dim3(256), 0, stream>>>(alpha, rowsum, u, colsum);
        col_pass<<<dim3(4, 32), dim3(256), 0, stream>>>(Kb, u, colsum);
        finish_v<<<dim3(32), dim3(256), 0, stream>>>(beta, colsum, v, vb, rowsum);
    }
    finalize<<<dim3(32), dim3(256), 0, stream>>>(u, v, eps, out);
}
